// TVTSBERT_76776835383517
// MI455X (gfx1250) — matrix operand order repaired and checked by reading, not compiled, not run on hardware
//
#include <hip/hip_runtime.h>
#include <hip/hip_bf16.h>

typedef _Float16 v16h __attribute__((ext_vector_type(16)));
typedef _Float16 v8h  __attribute__((ext_vector_type(8)));
typedef float    v8f  __attribute__((ext_vector_type(8)));

#define B_  32
#define S_  512
#define F_  10
#define D_  768
#define H_  12
#define L_  12
#define DK_ 64
#define FF_ 3072
#define M_  (B_ * S_)   // 16384 token rows

static __device__ __forceinline__ v8f zero8() {
  v8f z;
  #pragma unroll
  for (int i = 0; i < 8; ++i) z[i] = 0.0f;
  return z;
}

// Build a 16-half WMMA fragment from two contiguous 16B chunks.
static __device__ __forceinline__ v16h make_frag(const _Float16* p0, const _Float16* p1) {
  v8h lo = *(const v8h*)p0;
  v8h hi = *(const v8h*)p1;
  v16h f;
  #pragma unroll
  for (int e = 0; e < 8; ++e) { f[e] = lo[e]; f[e + 8] = hi[e]; }
  return f;
}

static __device__ __forceinline__ v8f wmma_f16(v16h a, v16h b, v8f c) {
  return __builtin_amdgcn_wmma_f32_16x16x32_f16(false, a, false, b, (short)0, c, false, false);
}

// Async global->LDS 16B copy (tracked by ASYNCcnt, in-order per wave).
// LDS byte offset = low 32 bits of the generic pointer (LDS aperture rule).
static __device__ __forceinline__ void async_copy_b128(const _Float16* lds_dst,
                                                       const _Float16* gsrc) {
  unsigned int       l = (unsigned int)(uintptr_t)(const void*)lds_dst;
  unsigned long long g = (unsigned long long)(uintptr_t)(const void*)gsrc;
  asm volatile("global_load_async_to_lds_b128 %0, %1, off"
               :: "v"(l), "v"(g) : "memory");
}
static __device__ __forceinline__ void async_wait0() {
  asm volatile("s_wait_asynccnt 0x0" ::: "memory");
}
// Leave the 8 most recent async ops outstanding (completes the older tile).
static __device__ __forceinline__ void async_wait8() {
  asm volatile("s_wait_asynccnt 0x8" ::: "memory");
}

// ---------------- embedding: h = x @ w_emb + b_emb, *sqrt(D), + pe ----------------
__global__ void embed_kernel(const float* __restrict__ x, const float* __restrict__ wemb,
                             const float* __restrict__ bemb, const float* __restrict__ pe,
                             float* __restrict__ h) {
  int idx = blockIdx.x * blockDim.x + threadIdx.x;
  int m = idx / D_;
  int n = idx - m * D_;
  int ss = m & (S_ - 1);
  float acc = bemb[n];
  #pragma unroll
  for (int f = 0; f < F_; ++f) acc += x[(size_t)m * F_ + f] * wemb[f * D_ + n];
  h[idx] = acc * 27.712812921102035f + pe[ss * D_ + n];   // sqrt(768)
}

// ---------------- per-layer weight prep: W_f32[K,N] -> Wt_f16[N,K] (LDS tiled) ----
__global__ void wconv_kernel(const float* __restrict__ W, _Float16* __restrict__ Wt,
                             int K, int N) {
  __shared__ _Float16 t[64 * 65];
  const int ktiles = K >> 6;
  const int kt = blockIdx.x % ktiles;
  const int nt = blockIdx.x / ktiles;
  const int k0 = kt * 64, n0 = nt * 64;
  #pragma unroll
  for (int i = 0; i < 16; ++i) {
    int idx = threadIdx.x + i * 256;
    int r = idx >> 6, c = idx & 63;                       // r: k-local, c: n-local
    t[c * 65 + r] = (_Float16)W[(size_t)(k0 + r) * N + n0 + c];
  }
  __syncthreads();
  #pragma unroll
  for (int i = 0; i < 16; ++i) {
    int idx = threadIdx.x + i * 256;
    int r = idx >> 6, c = idx & 63;                       // r: n-local, c: k-local
    Wt[(size_t)(n0 + r) * K + k0 + c] = t[r * 65 + c];
  }
}

// ---------------- LayerNorm (torch-style: unbiased var, /(std+eps)), f32 -> f16 ---
__global__ void ln_kernel(const float* __restrict__ h, const float* __restrict__ ga,
                          const float* __restrict__ be, _Float16* __restrict__ y) {
  __shared__ float red[8];
  const int row = blockIdx.x;
  const int tid = threadIdx.x;
  const float* hr = h + (size_t)row * D_;
  float x0 = hr[tid], x1 = hr[tid + 256], x2 = hr[tid + 512];
  float s = x0 + x1 + x2;
  #pragma unroll
  for (int off = 16; off > 0; off >>= 1) s += __shfl_xor(s, off, 32);
  if ((tid & 31) == 0) red[tid >> 5] = s;
  __syncthreads();
  float tot = 0.0f;
  #pragma unroll
  for (int w = 0; w < 8; ++w) tot += red[w];
  float mean = tot * (1.0f / 768.0f);
  float d0 = x0 - mean, d1 = x1 - mean, d2 = x2 - mean;
  float ss2 = d0 * d0 + d1 * d1 + d2 * d2;
  #pragma unroll
  for (int off = 16; off > 0; off >>= 1) ss2 += __shfl_xor(ss2, off, 32);
  __syncthreads();
  if ((tid & 31) == 0) red[tid >> 5] = ss2;
  __syncthreads();
  float tss = 0.0f;
  #pragma unroll
  for (int w = 0; w < 8; ++w) tss += red[w];
  float inv = 1.0f / (sqrtf(tss * (1.0f / 767.0f)) + 1e-6f);
  _Float16* yr = y + (size_t)row * D_;
  yr[tid]       = (_Float16)(ga[tid]       * d0 * inv + be[tid]);
  yr[tid + 256] = (_Float16)(ga[tid + 256] * d1 * inv + be[tid + 256]);
  yr[tid + 512] = (_Float16)(ga[tid + 512] * d2 * inv + be[tid + 512]);
}

// ---------------- GEMM: C[M,N] = A_f16[M,K] @ Wt_f16[N,K]^T + bias ---------------
// BM=128, BN=128, BK=64; double-buffered async LDS staging; 16 WMMAs per round.
// EPI 0: f16 out scattered to [B,H,S,DK] head layout (QKV projections)
// EPI 1: f32 residual accumulate into h[M, Ntot]  (O-proj / FF2)
// EPI 2: f16 out = gelu(acc + bias) at [M, Ntot]  (FF1)
template <int EPI>
__global__ void gemm_kernel(const _Float16* __restrict__ A, const _Float16* __restrict__ Wt,
                            const float* __restrict__ bias, void* __restrict__ outp,
                            int Ktot, int Ntot, int Ntiles) {
  __shared__ _Float16 As[2][128 * 64];   // [m][k]  2 x 16KB
  __shared__ _Float16 Ws[2][128 * 64];   // [n][k]  2 x 16KB
  const int tid  = threadIdx.x;
  const int lane = tid & 31;
  const int wave = tid >> 5;
  const int wm = wave >> 1, wn = wave & 1;
  const int mtile = blockIdx.x / Ntiles;
  const int ntile = blockIdx.x % Ntiles;
  const int m0 = mtile * 128;
  const int n0 = ntile * 128;
  const int ln16 = lane & 15;
  const int hi   = lane >> 4;

  // per-thread staging assignment: 4 chunks of 8 halves for each tile
  const int srow = (tid * 4) >> 3;        // tid/2
  const int skc0 = ((tid * 4) & 7) * 8;   // 0 or 32

  v8f acc[2][4];
  #pragma unroll
  for (int i = 0; i < 2; ++i)
    #pragma unroll
    for (int j = 0; j < 4; ++j) acc[i][j] = zero8();

  const int nk = Ktot >> 6;

  // prologue: stage tile 0 into buffer 0
  #pragma unroll
  for (int c = 0; c < 4; ++c) {
    int row = srow + ((skc0 + c * 8) >> 6);       // stays srow (skc0+c*8 < 64)
    int kc  = (skc0 + c * 8) & 63;
    async_copy_b128(&As[0][row * 64 + kc], A  + (size_t)(m0 + row) * Ktot + kc);
    async_copy_b128(&Ws[0][row * 64 + kc], Wt + (size_t)(n0 + row) * Ktot + kc);
  }

  for (int it = 0; it < nk; ++it) {
    const int cur = it & 1;
    __syncthreads();   // all waves done reading buffer (it+1)&1 from iteration it-1
    if (it + 1 < nk) {
      const int nxt = (it + 1) & 1;
      const int kb  = (it + 1) << 6;
      #pragma unroll
      for (int c = 0; c < 4; ++c) {
        int kc  = skc0 + ((c * 8) & 31) + ((c >> 2) << 5); // kc in {skc0, skc0+8, skc0+16, skc0+24}
        kc = skc0 + c * 8;
        async_copy_b128(&As[nxt][srow * 64 + kc], A  + (size_t)(m0 + srow) * Ktot + kb + kc);
        async_copy_b128(&Ws[nxt][srow * 64 + kc], Wt + (size_t)(n0 + srow) * Ktot + kb + kc);
      }
      async_wait8();   // oldest 8 (current tile) complete; newest 8 still in flight
    } else {
      async_wait0();
    }
    __syncthreads();   // current tile visible to all waves

    #pragma unroll
    for (int ks = 0; ks < 64; ks += 32) {
      v16h af[2], bf[4];
      #pragma unroll
      for (int i = 0; i < 2; ++i) {
        const _Float16* base = &As[cur][(wm * 32 + i * 16 + ln16) * 64 + ks];
        af[i] = make_frag(base + hi * 8, base + 16 + hi * 8);
      }
      #pragma unroll
      for (int j = 0; j < 4; ++j) {
        const _Float16* base = &Ws[cur][(wn * 64 + j * 16 + ln16) * 64 + ks];
        bf[j] = make_frag(base + hi * 8, base + 16 + hi * 8);
      }
      #pragma unroll
      for (int i = 0; i < 2; ++i)
        #pragma unroll
        for (int j = 0; j < 4; ++j)
          acc[i][j] = wmma_f16(af[i], bf[j], acc[i][j]);
    }
  }

  #pragma unroll
  for (int i = 0; i < 2; ++i) {
    #pragma unroll
    for (int j = 0; j < 4; ++j) {
      int n = n0 + wn * 64 + j * 16 + ln16;
      float bv = bias[n];
      #pragma unroll
      for (int r = 0; r < 8; ++r) {
        int m = m0 + wm * 32 + i * 16 + hi * 8 + r;
        float val = acc[i][j][r] + bv;
        if (EPI == 0) {
          int bb = m / S_, ssq = m - bb * S_;
          int head = n / DK_, dk = n - head * DK_;
          ((_Float16*)outp)[((size_t)(bb * H_ + head) * S_ + ssq) * DK_ + dk] = (_Float16)val;
        } else if (EPI == 1) {
          float* hp = (float*)outp;
          size_t idx = (size_t)m * Ntot + n;
          hp[idx] = hp[idx] + val;
        } else {
          float xg = val;
          float g = 0.5f * xg *
                    (1.0f + tanhf(0.7978845608028654f * (xg + 0.044715f * xg * xg * xg)));
          ((_Float16*)outp)[(size_t)m * Ntot + n] = (_Float16)g;
        }
      }
    }
  }
}

// ---------------- flash attention: one block per (b, head, 128-query tile) -------
__global__ void attn_kernel(const _Float16* __restrict__ q, const _Float16* __restrict__ k,
                            const _Float16* __restrict__ v, const int* __restrict__ mask,
                            _Float16* __restrict__ o) {
  __shared__ _Float16 Ks[32 * 64];       // [kk][d]
  __shared__ _Float16 Vt[64 * 32];       // [d][kk]
  __shared__ _Float16 Ps[8][16 * 32];    // per-wave P scratch
  const int tid  = threadIdx.x;
  const int lane = tid & 31;
  const int wave = tid >> 5;
  const int ln16 = lane & 15;
  const int hi   = lane >> 4;
  const int bh   = blockIdx.x >> 2;      // S/128 = 4 query tiles
  const int qt   = blockIdx.x & 3;
  const int bb   = bh / H_;
  const int head = bh - bb * H_;
  const int q0   = qt * 128 + wave * 16;

  // Q fragments (2 x 16x32 A-frags covering DK=64), pre-scaled by 1/sqrt(64)
  const _Float16* qrow = q + ((size_t)bh * S_ + q0 + ln16) * DK_;
  v16h qa0 = make_frag(qrow + hi * 8,      qrow + 16 + hi * 8);
  v16h qa1 = make_frag(qrow + 32 + hi * 8, qrow + 48 + hi * 8);
  #pragma unroll
  for (int e = 0; e < 16; ++e) {
    qa0[e] = qa0[e] * (_Float16)0.125f;
    qa1[e] = qa1[e] * (_Float16)0.125f;
  }

  float mrow[8], lrow[8];
  v8f oacc[4];
  #pragma unroll
  for (int r = 0; r < 8; ++r) { mrow[r] = -3.0e38f; lrow[r] = 0.0f; }
  #pragma unroll
  for (int t = 0; t < 4; ++t) oacc[t] = zero8();

  for (int jt = 0; jt < S_ / 32; ++jt) {
    __syncthreads();
    {   // stage K tile async (row-major); V tile transposed via VGPR bounce
      int row = tid >> 3;
      int dc  = (tid & 7) * 8;
      async_copy_b128(Ks + row * 64 + dc,
                      k + ((size_t)bh * S_ + jt * 32 + row) * DK_ + dc);
      v8h vv = *(const v8h*)(v + ((size_t)bh * S_ + jt * 32 + row) * DK_ + dc);
      #pragma unroll
      for (int e = 0; e < 8; ++e) Vt[(dc + e) * 32 + row] = vv[e];
      async_wait0();
    }
    __syncthreads();

    // scores for 32 keys: two 16x16 accumulators, K-dim 64 = 2 WMMAs each
    v8f sc[2];
    #pragma unroll
    for (int c = 0; c < 2; ++c) {
      const _Float16* kbase = Ks + (c * 16 + ln16) * 64;
      v16h b0 = make_frag(kbase + hi * 8,      kbase + 16 + hi * 8);
      v16h b1 = make_frag(kbase + 32 + hi * 8, kbase + 48 + hi * 8);
      v8f s = zero8();
      s = wmma_f16(qa0, b0, s);
      s = wmma_f16(qa1, b1, s);
      sc[c] = s;
    }

    // key mask (broadcast over queries and heads)
    int key0 = jt * 32 + ln16;
    int mk0 = mask[bb * S_ + key0];
    int mk1 = mask[bb * S_ + key0 + 16];
    if (!mk0) {
      #pragma unroll
      for (int r = 0; r < 8; ++r) sc[0][r] = -1.0e9f;
    }
    if (!mk1) {
      #pragma unroll
      for (int r = 0; r < 8; ++r) sc[1][r] = -1.0e9f;
    }

    // online softmax update (row stats via half-wave shuffles)
    float p0[8], p1[8], corr[8];
    #pragma unroll
    for (int r = 0; r < 8; ++r) {
      float x = fmaxf(sc[0][r], sc[1][r]);
      x = fmaxf(x, __shfl_xor(x, 1, 16));
      x = fmaxf(x, __shfl_xor(x, 2, 16));
      x = fmaxf(x, __shfl_xor(x, 4, 16));
      x = fmaxf(x, __shfl_xor(x, 8, 16));
      float mn = fmaxf(mrow[r], x);
      corr[r] = __expf(mrow[r] - mn);
      mrow[r] = mn;
      p0[r] = __expf(sc[0][r] - mn);
      p1[r] = __expf(sc[1][r] - mn);
      float ps = p0[r] + p1[r];
      ps += __shfl_xor(ps, 1, 16);
      ps += __shfl_xor(ps, 2, 16);
      ps += __shfl_xor(ps, 4, 16);
      ps += __shfl_xor(ps, 8, 16);
      lrow[r] = lrow[r] * corr[r] + ps;
    }
    #pragma unroll
    for (int t = 0; t < 4; ++t)
      #pragma unroll
      for (int r = 0; r < 8; ++r) oacc[t][r] *= corr[r];

    // bounce P (16x32) through per-wave LDS to get A-fragment layout
    _Float16* pw = Ps[wave];
    #pragma unroll
    for (int r = 0; r < 8; ++r) {
      pw[(hi * 8 + r) * 32 + ln16]      = (_Float16)p0[r];
      pw[(hi * 8 + r) * 32 + 16 + ln16] = (_Float16)p1[r];
    }
    asm volatile("s_wait_dscnt 0" ::: "memory");  // wave-local LDS RAW fence
    const _Float16* pr = pw + ln16 * 32;
    v16h pa = make_frag(pr + hi * 8, pr + 16 + hi * 8);

    // O += P @ V  (4 d-subtiles of 16)
    #pragma unroll
    for (int t = 0; t < 4; ++t) {
      const _Float16* vb = Vt + (t * 16 + ln16) * 32;
      v16h bfv = make_frag(vb + hi * 8, vb + 16 + hi * 8);
      oacc[t] = wmma_f16(pa, bfv, oacc[t]);
    }
  }

  // finalize: O /= l, write to [B,S,D] layout
  #pragma unroll
  for (int t = 0; t < 4; ++t) {
    #pragma unroll
    for (int r = 0; r < 8; ++r) {
      float val = oacc[t][r] / fmaxf(lrow[r], 1e-30f);
      size_t idx = ((size_t)bb * S_ + q0 + hi * 8 + r) * D_ + head * DK_ + t * 16 + ln16;
      o[idx] = (_Float16)val;
    }
  }
}

// -------------------------------- host launch ------------------------------------
extern "C" void kernel_launch(void* const* d_in, const int* in_sizes, int n_in,
                              void* d_out, int out_size, void* d_ws, size_t ws_size,
                              hipStream_t stream) {
  (void)in_sizes; (void)n_in; (void)out_size; (void)ws_size;
  const float* x    = (const float*)d_in[0];
  const int*   mask = (const int*)d_in[1];
  const float* wemb = (const float*)d_in[2];
  const float* bemb = (const float*)d_in[3];
  const float* pe   = (const float*)d_in[4];
  const float* wq = (const float*)d_in[5];
  const float* bq = (const float*)d_in[6];
  const float* wk = (const float*)d_in[7];
  const float* bk = (const float*)d_in[8];
  const float* wv = (const float*)d_in[9];
  const float* bv = (const float*)d_in[10];
  const float* wo = (const float*)d_in[11];
  const float* bo = (const float*)d_in[12];
  const float* w1 = (const float*)d_in[13];
  const float* b1 = (const float*)d_in[14];
  const float* w2 = (const float*)d_in[15];
  const float* b2 = (const float*)d_in[16];
  const float* l1a = (const float*)d_in[17];
  const float* l1b = (const float*)d_in[18];
  const float* l2a = (const float*)d_in[19];
  const float* l2b = (const float*)d_in[20];

  float* h = (float*)d_out;                       // persistent residual stream
  char* ws = (char*)d_ws;
  const size_t ACT = (size_t)M_ * D_ * sizeof(_Float16);  // 24 MB
  _Float16* y   = (_Float16*)ws;                  // LN output
  _Float16* qb  = (_Float16*)(ws + 1 * ACT);
  _Float16* kb  = (_Float16*)(ws + 2 * ACT);
  _Float16* vb_ = (_Float16*)(ws + 3 * ACT);
  _Float16* ob  = (_Float16*)(ws + 4 * ACT);
  _Float16* ff  = (_Float16*)(ws + 1 * ACT);      // overlays q/k/v/o (4*ACT = M*FF*2)

  // per-layer f16 transposed weights
  const size_t DD2 = (size_t)D_ * D_  * sizeof(_Float16);
  const size_t DF2 = (size_t)D_ * FF_ * sizeof(_Float16);
  char* wb = ws + 5 * ACT;
  _Float16* wtq = (_Float16*)(wb);
  _Float16* wtk = (_Float16*)(wb + 1 * DD2);
  _Float16* wtv = (_Float16*)(wb + 2 * DD2);
  _Float16* wto = (_Float16*)(wb + 3 * DD2);
  _Float16* wt1 = (_Float16*)(wb + 4 * DD2);
  _Float16* wt2 = (_Float16*)(wb + 4 * DD2 + DF2);

  dim3 blk(256);
  embed_kernel<<<(M_ * D_) / 256, blk, 0, stream>>>(x, wemb, bemb, pe, h);

  const int gD   = (M_ / 128) * (D_ / 128);       // 128*6
  const int gFF  = (M_ / 128) * (FF_ / 128);      // 128*24
  const int gWdd = (D_ / 64) * (D_ / 64);         // 144
  const int gWdf = (D_ / 64) * (FF_ / 64);        // 576

  for (int l = 0; l < L_; ++l) {
    wconv_kernel<<<gWdd, blk, 0, stream>>>(wq + (size_t)l * D_ * D_,  wtq, D_,  D_);
    wconv_kernel<<<gWdd, blk, 0, stream>>>(wk + (size_t)l * D_ * D_,  wtk, D_,  D_);
    wconv_kernel<<<gWdd, blk, 0, stream>>>(wv + (size_t)l * D_ * D_,  wtv, D_,  D_);
    wconv_kernel<<<gWdd, blk, 0, stream>>>(wo + (size_t)l * D_ * D_,  wto, D_,  D_);
    wconv_kernel<<<gWdf, blk, 0, stream>>>(w1 + (size_t)l * D_ * FF_, wt1, D_,  FF_);
    wconv_kernel<<<gWdf, blk, 0, stream>>>(w2 + (size_t)l * FF_ * D_, wt2, FF_, D_);

    ln_kernel<<<M_, blk, 0, stream>>>(h, l1a + l * D_, l1b + l * D_, y);
    gemm_kernel<0><<<gD, blk, 0, stream>>>(y, wtq, bq + l * D_, qb,  D_, D_, D_ / 128);
    gemm_kernel<0><<<gD, blk, 0, stream>>>(y, wtk, bk + l * D_, kb,  D_, D_, D_ / 128);
    gemm_kernel<0><<<gD, blk, 0, stream>>>(y, wtv, bv + l * D_, vb_, D_, D_, D_ / 128);
    attn_kernel<<<B_ * H_ * (S_ / 128), blk, 0, stream>>>(qb, kb, vb_, mask, ob);
    gemm_kernel<1><<<gD, blk, 0, stream>>>(ob, wto, bo + l * D_, h, D_, D_, D_ / 128);
    ln_kernel<<<M_, blk, 0, stream>>>(h, l2a + l * D_, l2b + l * D_, y);
    gemm_kernel<2><<<gFF, blk, 0, stream>>>(y,  wt1, b1 + l * FF_, ff, D_, FF_, FF_ / 128);
    gemm_kernel<1><<<gD, blk, 0, stream>>>(ff, wt2, b2 + l * D_, h, FF_, D_, D_ / 128);
  }
}
